// FARGAN_17016660426804
// MI455X (gfx1250) — matrix operand order repaired and checked by reading, not compiled, not run on hardware
//
#include <hip/hip_runtime.h>
#include <hip/hip_bf16.h>

// ---------------------------------------------------------------------------
// FARGAN forward for MI455X (gfx1250): bf16 WMMA everywhere.
// B=256, T=68, NBF=64, FEAT=20, COND=256, SUB=40, NSUB=4, FRAME=160, PEMB=64
// ---------------------------------------------------------------------------

typedef __attribute__((ext_vector_type(16))) __bf16 v16bf;
typedef __attribute__((ext_vector_type(8)))  __bf16 v8bf;
typedef __attribute__((ext_vector_type(8)))  float  v8f;

__device__ __forceinline__ __bf16 f2bf(float f) {
  unsigned u = __builtin_bit_cast(unsigned, f);
  u += 0x7FFFu + ((u >> 16) & 1u);           // round-to-nearest-even
  unsigned short h = (unsigned short)(u >> 16);
  return __builtin_bit_cast(__bf16, h);
}
__device__ __forceinline__ float bf2f(__bf16 b) {
  unsigned u = ((unsigned)__builtin_bit_cast(unsigned short, b)) << 16;
  return __builtin_bit_cast(float, u);
}

// Branch-free activations on v_exp_f32 / v_rcp_f32 (TRANS ops co-execute with WMMA).
#define LOG2E 1.4426950408889634f
__device__ __forceinline__ float fast_sigm(float x) {
  return __builtin_amdgcn_rcpf(1.f + __builtin_amdgcn_exp2f(-x * LOG2E));
}
__device__ __forceinline__ float fast_tanh(float x) {
  // 1 - 2/(e^{2x}+1); saturates to +/-1 via exp2 overflow/underflow, no branches
  return 1.f - 2.f * __builtin_amdgcn_rcpf(__builtin_amdgcn_exp2f(2.f * x * LOG2E) + 1.f);
}

__device__ __forceinline__ v8f wmma_bf16(v16bf a, v16bf b, v8f c) {
  // 8 args: (neg_a, A, neg_b, B, c_mod, C, reuse_a, reuse_b)
  return __builtin_amdgcn_wmma_f32_16x16x32_bf16(false, a, false, b, (short)0, c,
                                                 false, false);
}

// A fragment (16x32 bf16) from LDS row-major [16][ld].
// ISA layout: lanes 0-15 rows 0-15 K={0..7,16..23}; lanes 16-31 same rows K={8..15,24..31}
__device__ __forceinline__ v16bf load_a_lds(const __bf16* base, int ld, int k0, int lane) {
  int r = lane & 15, g = lane >> 4;
  const __bf16* p = base + r * ld + k0 + g * 8;
  v8bf lo = *(const v8bf*)(p);
  v8bf hi = *(const v8bf*)(p + 16);
  union { v16bf v; v8bf h[2]; } u;
  u.h[0] = lo; u.h[1] = hi;
  return u.v;
}

// A fragment from global f32 row (per-lane row base), vector loads + convert.
__device__ __forceinline__ v16bf load_a_g32(const float* rowbase, int k0, int g) {
  const float4* p0 = (const float4*)(rowbase + k0 + g * 8);
  const float4* p1 = (const float4*)(rowbase + k0 + 16 + g * 8);
  float4 x0 = p0[0], x1 = p0[1], y0 = p1[0], y1 = p1[1];
  v16bf a;
  a[0] = f2bf(x0.x);  a[1] = f2bf(x0.y);  a[2] = f2bf(x0.z);  a[3] = f2bf(x0.w);
  a[4] = f2bf(x1.x);  a[5] = f2bf(x1.y);  a[6] = f2bf(x1.z);  a[7] = f2bf(x1.w);
  a[8] = f2bf(y0.x);  a[9] = f2bf(y0.y);  a[10] = f2bf(y0.z); a[11] = f2bf(y0.w);
  a[12] = f2bf(y1.x); a[13] = f2bf(y1.y); a[14] = f2bf(y1.z); a[15] = f2bf(y1.w);
  return a;
}

// B fragment (32x16) from packed weights: [tile][lane][16] bf16, 32B aligned.
__device__ __forceinline__ v16bf load_b_pack(const __bf16* pack, int KT, int nTile,
                                             int kTile, int lane) {
  const __bf16* p = pack + (((size_t)nTile * KT + kTile) * 32 + lane) * 16;
  return *(const v16bf*)p;
}

// C layout: element j -> row m = (lane>>4)*8 + j, col n = nTile*16 + (lane&15)
__device__ __forceinline__ void store_tanh_lds(v8f acc, __bf16* dst, int dld, int nt,
                                               int lane) {
  int r = lane & 15, g = lane >> 4;
  int c = nt * 16 + r;
#pragma unroll
  for (int j = 0; j < 8; ++j) dst[(g * 8 + j) * dld + c] = f2bf(fast_tanh(acc[j]));
}
__device__ __forceinline__ void store_glu_lds(v8f acc, const __bf16* src, int sld,
                                              __bf16* dst, int dld, int nt, int lane) {
  int r = lane & 15, g = lane >> 4;
  int c = nt * 16 + r;
#pragma unroll
  for (int j = 0; j < 8; ++j) {
    float x = bf2f(src[(g * 8 + j) * sld + c]);
    dst[(g * 8 + j) * dld + c] = f2bf(x * fast_sigm(acc[j]));
  }
}

// Two-column-tile GEMM from LDS activations (N=256 -> tiles wave and wave+8).
__device__ __forceinline__ void gemm2(const __bf16* A, int ld, int KT, const __bf16* Bp,
                                      int wave, int lane, v8f& acc0, v8f& acc1) {
  acc0 = (v8f){0, 0, 0, 0, 0, 0, 0, 0};
  acc1 = (v8f){0, 0, 0, 0, 0, 0, 0, 0};
  for (int kt = 0; kt < KT; ++kt) {
    v16bf a = load_a_lds(A, ld, kt * 32, lane);
    v16bf b0 = load_b_pack(Bp, KT, wave, kt, lane);
    v16bf b1 = load_b_pack(Bp, KT, wave + 8, kt, lane);
    acc0 = wmma_bf16(a, b0, acc0);
    acc1 = wmma_bf16(a, b1, acc1);
  }
}

// ---------------------------------------------------------------------------
// Weight repack: f32 (N x K logical) -> bf16 WMMA B-fragment order.
// mode 0: row-major src[n*realK + k]; mode 1: conv (O,I,3): src[n*768 + (k%256)*3 + k/256]
// ---------------------------------------------------------------------------
__global__ void pack_w(const float* __restrict__ src, __bf16* __restrict__ dst, int N,
                       int K, int realN, int realK, int mode) {
  int NT = N >> 4, KT = K >> 5;
  long total = (long)NT * KT * 512;
  long i = (long)blockIdx.x * blockDim.x + threadIdx.x;
  if (i >= total) return;
  int e = (int)(i & 15);
  int lane = (int)((i >> 4) & 31);
  long tile = i >> 9;
  int kTile = (int)(tile % KT);
  int nTile = (int)(tile / KT);
  int n = nTile * 16 + (lane & 15);
  int k = kTile * 32 + (lane >> 4) * 8 + (e & 7) + (e >> 3) * 16;
  float v = 0.f;
  if (n < realN && k < realK) {
    long idx;
    if (mode == 0) idx = (long)n * realK + k;
    else idx = (long)n * 768 + (long)(k & 255) * 3 + (k >> 8);
    v = src[idx];
  }
  dst[i] = f2bf(v);
}

// ---------------------------------------------------------------------------
// Build dense1 input: rows (b,t), cols [feat(20) | pembed[period](64) | pad(12)]
// ---------------------------------------------------------------------------
__global__ void build_ain(const float* __restrict__ features, const int* __restrict__ period,
                          const float* __restrict__ pembed, float* __restrict__ Ain) {
  long i = (long)blockIdx.x * blockDim.x + threadIdx.x;
  if (i >= 256L * 68 * 96) return;
  int c = (int)(i % 96);
  long bt = i / 96;
  int t = (int)(bt % 68);
  int b = (int)(bt / 68);
  float v = 0.f;
  if (c < 20) v = features[((long)b * 68 + t) * 20 + c];
  else if (c < 84) v = pembed[(long)period[b * 68 + t] * 64 + (c - 20)];
  Ain[i] = v;
}

// ---------------------------------------------------------------------------
// Generic tiled GEMM: C[b,t,:] = tanh(A[b,t,:] @ Bp^T), A f32 global, Bp packed bf16.
// RPB (output rows per batch) and K are compile-time: division-by-constant + unroll.
// Each wave computes one 16x16 output tile.
// ---------------------------------------------------------------------------
template <int RPB, int K>
__global__ __launch_bounds__(256) void gemm_rt(const float* __restrict__ A, int bsA,
                                               int lda, const __bf16* __restrict__ Bp,
                                               float* __restrict__ C, int bsC, int ldc,
                                               int M) {
  constexpr int NT = 16;   // N = 256
  constexpr int KT = K / 32;
  int wave = threadIdx.x >> 5, lane = threadIdx.x & 31;
  long MT = M >> 4;
  long tileId = (long)blockIdx.x * 8 + wave;
  if (tileId >= MT * (long)NT) return;
  int nTile = (int)(tileId & (NT - 1));
  long mTile = tileId >> 4;
  int r = lane & 15, g = lane >> 4;
  // A row for this lane (division by compile-time constant -> magic multiply)
  unsigned row = (unsigned)(mTile * 16) + r;
  unsigned b = row / RPB, t = row % RPB;
  const float* arow = A + (long)b * bsA + (long)t * lda;
  v8f acc = {0, 0, 0, 0, 0, 0, 0, 0};
#pragma unroll
  for (int kt = 0; kt < KT; ++kt) {
    v16bf a = load_a_g32(arow, kt * 32, g);
    v16bf bf = load_b_pack(Bp, KT, nTile, kt, lane);
    acc = wmma_bf16(a, bf, acc);
  }
  int n = nTile * 16 + r;
  // first output row of this lane's half; tile spans at most one batch boundary
  unsigned row0 = (unsigned)(mTile * 16) + g * 8;
  unsigned ob0 = row0 / RPB, ot0 = row0 % RPB;
#pragma unroll
  for (int j = 0; j < 8; ++j) {
    unsigned ob = ob0, ot = ot0 + j;
    if (ot >= (unsigned)RPB) { ob = ob0 + 1; ot -= RPB; }
    C[(long)ob * bsC + (long)ot * ldc + n] = fast_tanh(acc[j]);
  }
}

// ---------------------------------------------------------------------------
// Phase embedding: threefry2x32(key=[0,42]) reproduces jax.random.uniform(key(42)).
// Writes bf16 [b][s*80 + {cos:0..39, sin:40..79}], s=0..255.
// ---------------------------------------------------------------------------
__device__ __forceinline__ unsigned rotl32(unsigned x, int d) {
  return (x << d) | (x >> (32 - d));
}
__device__ void threefry2x32(unsigned k0, unsigned k1, unsigned x0, unsigned x1,
                             unsigned* o0, unsigned* o1) {
  unsigned ks2 = k0 ^ k1 ^ 0x1BD11BDAu;
  const int ra[4] = {13, 15, 26, 6}, rb[4] = {17, 29, 16, 24};
  x0 += k0; x1 += k1;
#pragma unroll
  for (int i = 0; i < 4; ++i) { x0 += x1; x1 = rotl32(x1, ra[i]); x1 ^= x0; }
  x0 += k1; x1 += ks2 + 1;
#pragma unroll
  for (int i = 0; i < 4; ++i) { x0 += x1; x1 = rotl32(x1, rb[i]); x1 ^= x0; }
  x0 += ks2; x1 += k0 + 2;
#pragma unroll
  for (int i = 0; i < 4; ++i) { x0 += x1; x1 = rotl32(x1, ra[i]); x1 ^= x0; }
  x0 += k0; x1 += k1 + 3;
#pragma unroll
  for (int i = 0; i < 4; ++i) { x0 += x1; x1 = rotl32(x1, rb[i]); x1 ^= x0; }
  x0 += k1; x1 += ks2 + 4;
#pragma unroll
  for (int i = 0; i < 4; ++i) { x0 += x1; x1 = rotl32(x1, ra[i]); x1 ^= x0; }
  x0 += ks2; x1 += k0 + 5;
  *o0 = x0; *o1 = x1;
}

__global__ void phase_kernel(const int* __restrict__ period, __bf16* __restrict__ phase) {
  __shared__ float w0s[64], cums[64];
  int b = blockIdx.x;
  if (threadIdx.x == 0) {
    unsigned x0, x1, o0, o1;
    if (b < 128) { x0 = (unsigned)b; x1 = (unsigned)(b + 128); }
    else { x0 = (unsigned)(b - 128); x1 = (unsigned)b; }
    threefry2x32(0u, 42u, x0, x1, &o0, &o1);
    unsigned bits = (b < 128) ? o0 : o1;
    float u = __builtin_bit_cast(float, (bits >> 9) | 0x3F800000u) - 1.0f;
    const float TWO_PI = 6.2831853071795864769f;
    float shift0 = TWO_PI * u / 160.f;
    float run = 0.f;
    for (int t = 0; t < 64; ++t) {
      float w = TWO_PI / (float)period[b * 68 + 3 + t];
      float wstep = (t == 0) ? shift0 : w0s[t - 1];
      w0s[t] = w;
      run += 160.f * wstep;
      cums[t] = run;
    }
  }
  __syncthreads();
  for (int i = threadIdx.x; i < 10240; i += blockDim.x) {
    int t = i / 160, j = i % 160;
    float e = cums[t] + w0s[t] * (float)j;
    int s = i / 40, q = i % 40;
    phase[(long)b * 20480 + s * 80 + q] = f2bf(cosf(e));
    phase[(long)b * 20480 + s * 80 + 40 + q] = f2bf(sinf(e));
  }
}

// ---------------------------------------------------------------------------
// Persistent recurrent kernel: 16 workgroups x 16 batch rows, 256 steps.
// LDS: act[16][384] bf16, act2[16][256] bf16, hs[3][16][256] bf16, exc ring f32.
// ---------------------------------------------------------------------------
__global__ __launch_bounds__(256) void fargan_rnn(
    const float* __restrict__ cond, const __bf16* __restrict__ phase,
    const int* __restrict__ period, const __bf16* Wd1, const __bf16* Gd1,
    const __bf16* Wd2, const __bf16* Gd2, const __bf16* Wih1, const __bf16* Whh1,
    const __bf16* Gg1, const __bf16* Wih2, const __bf16* Whh2, const __bf16* Gg2,
    const __bf16* Wih3, const __bf16* Whh3, const __bf16* Gg3, const __bf16* Wout,
    float* __restrict__ out) {
  __shared__ __align__(16) __bf16 act[16 * 384];
  __shared__ __align__(16) __bf16 act2[16 * 256];
  __shared__ __align__(16) __bf16 hs[3][16 * 256];
  __shared__ __align__(16) float exc[16 * 256];

  const int tid = threadIdx.x;
  const int wave = tid >> 5, lane = tid & 31;
  const int rr = lane & 15, gg = lane >> 4;
  const int m0 = blockIdx.x * 16;

  for (int i = tid; i < 16 * 256; i += 256) {
    exc[i] = 0.f;
    hs[0][i] = f2bf(0.f); hs[1][i] = f2bf(0.f); hs[2][i] = f2bf(0.f);
  }
  for (int i = tid; i < 16 * 384; i += 256) act[i] = f2bf(0.f);
  __syncthreads();

  const __bf16* WihA[3] = {Wih1, Wih2, Wih3};
  const __bf16* WhhA[3] = {Whh1, Whh2, Whh3};
  const __bf16* GgA[3] = {Gg1, Gg2, Gg3};

  int pos = 0;
#pragma unroll 1
  for (int s = 0; s < 256; ++s) {
    const int frame = s >> 2;
    // ---- build step input: [cond(256) | prev_exc(40) | phase(80) | pad] ----
    for (int i = tid; i < 16 * 64; i += 256) {           // cond: float4 loads
      int row = i >> 6, c4 = (i & 63) * 4;
      float4 v = *(const float4*)&cond[((long)(m0 + row) * 64 + frame) * 256 + c4];
      __bf16* d = &act[row * 384 + c4];
      d[0] = f2bf(v.x); d[1] = f2bf(v.y); d[2] = f2bf(v.z); d[3] = f2bf(v.w);
    }
    for (int i = tid; i < 16 * 40; i += 256) {           // pitch-gather from exc ring
      int row = i / 40, j = i % 40;
      int pit = period[(m0 + row) * 68 + 3 + frame];
      int mx = pit > 40 ? pit : 40;
      act[row * 384 + 256 + j] = f2bf(exc[row * 256 + ((pos - mx + j) & 255)]);
    }
    for (int i = tid; i < 16 * 20; i += 256) {           // phase: 4x bf16 copies
      int row = i / 20, q4 = (i % 20) * 4;
      *(uint2*)&act[row * 384 + 296 + q4] =
          *(const uint2*)&phase[(long)(m0 + row) * 20480 + s * 80 + q4];
    }
    __syncthreads();

    v8f a0, a1;
    // tmp = glu(tanh(in @ Wd1), Gd1)
    gemm2(act, 384, 12, Wd1, wave, lane, a0, a1);
    __syncthreads();
    store_tanh_lds(a0, act2, 256, wave, lane);
    store_tanh_lds(a1, act2, 256, wave + 8, lane);
    __syncthreads();
    gemm2(act2, 256, 8, Gd1, wave, lane, a0, a1);
    __syncthreads();
    store_glu_lds(a0, act2, 256, act, 384, wave, lane);
    store_glu_lds(a1, act2, 256, act, 384, wave + 8, lane);
    __syncthreads();
    // tmp = glu(tanh(tmp @ Wd2), Gd2)
    gemm2(act, 384, 8, Wd2, wave, lane, a0, a1);
    __syncthreads();
    store_tanh_lds(a0, act2, 256, wave, lane);
    store_tanh_lds(a1, act2, 256, wave + 8, lane);
    __syncthreads();
    gemm2(act2, 256, 8, Gd2, wave, lane, a0, a1);
    __syncthreads();
    store_glu_lds(a0, act2, 256, act, 384, wave, lane);
    store_glu_lds(a1, act2, 256, act, 384, wave + 8, lane);
    __syncthreads();

    // ---- 3x (GRU + GLU) ----
#pragma unroll 1
    for (int gi = 0; gi < 3; ++gi) {
      __bf16* h = hs[gi];
      v8f gacc[12];
#pragma unroll
      for (int q = 0; q < 12; ++q) gacc[q] = (v8f){0, 0, 0, 0, 0, 0, 0, 0};
      for (int kt = 0; kt < 8; ++kt) {
        v16bf ax = load_a_lds(act, 384, kt * 32, lane);
        v16bf ah = load_a_lds(h, 256, kt * 32, lane);
#pragma unroll
        for (int j = 0; j < 2; ++j) {
          int ntb = wave + 8 * j;
#pragma unroll
          for (int gate = 0; gate < 3; ++gate) {
            v16bf bi = load_b_pack(WihA[gi], 8, gate * 16 + ntb, kt, lane);
            v16bf bh = load_b_pack(WhhA[gi], 8, gate * 16 + ntb, kt, lane);
            gacc[j * 6 + gate * 2 + 0] = wmma_bf16(ax, bi, gacc[j * 6 + gate * 2 + 0]);
            gacc[j * 6 + gate * 2 + 1] = wmma_bf16(ah, bh, gacc[j * 6 + gate * 2 + 1]);
          }
        }
      }
      __syncthreads();  // all whh reads of h done before updating h
#pragma unroll
      for (int j = 0; j < 2; ++j) {
        int c = (wave + 8 * j) * 16 + rr;
#pragma unroll
        for (int e = 0; e < 8; ++e) {
          int m = gg * 8 + e;
          float rg = fast_sigm(gacc[j * 6 + 0][e] + gacc[j * 6 + 1][e]);
          float zg = fast_sigm(gacc[j * 6 + 2][e] + gacc[j * 6 + 3][e]);
          float ng = fast_tanh(gacc[j * 6 + 4][e] + rg * gacc[j * 6 + 5][e]);
          float hold = bf2f(h[m * 256 + c]);
          h[m * 256 + c] = f2bf((1.f - zg) * ng + zg * hold);
        }
      }
      __syncthreads();
      // next input = glu(h, Gg)
      gemm2(h, 256, 8, GgA[gi], wave, lane, a0, a1);
      __syncthreads();
      store_glu_lds(a0, h, 256, act, 384, wave, lane);
      store_glu_lds(a1, h, 256, act, 384, wave + 8, lane);
      __syncthreads();
    }

    // ---- sig_out = tanh(glu(h3) @ Wout^T), write exc ring + output ----
    gemm2(act, 384, 8, Wout, wave, lane, a0, a1);
    __syncthreads();
    {
      int c = wave * 16 + rr;
      if (c < 40) {
#pragma unroll
        for (int e = 0; e < 8; ++e) {
          int m = gg * 8 + e;
          float v = fast_tanh(a0[e]);
          exc[m * 256 + ((pos + c) & 255)] = v;
          out[(long)(m0 + m) * 10240 + s * 40 + c] = v;
        }
      }
    }
    __syncthreads();
    pos = (pos + 40) & 255;
  }
}

// ---------------------------------------------------------------------------
extern "C" void kernel_launch(void* const* d_in, const int* in_sizes, int n_in,
                              void* d_out, int out_size, void* d_ws, size_t ws_size,
                              hipStream_t stream) {
  (void)in_sizes; (void)n_in; (void)out_size; (void)ws_size;
  const float* features = (const float*)d_in[0];
  const int* period = (const int*)d_in[1];
  const float* pembed = (const float*)d_in[3];
  const float* fdense1_w = (const float*)d_in[4];
  const float* fconv1_w = (const float*)d_in[5];
  const float* fconv2_w = (const float*)d_in[6];
  const float* fdense2_w = (const float*)d_in[7];
  const float* sig_dense1_w = (const float*)d_in[8];
  const float* sig_dense2_w = (const float*)d_in[9];
  const float* gru1_wih = (const float*)d_in[10];
  const float* gru1_whh = (const float*)d_in[11];
  const float* gru2_wih = (const float*)d_in[12];
  const float* gru2_whh = (const float*)d_in[13];
  const float* gru3_wih = (const float*)d_in[14];
  const float* gru3_whh = (const float*)d_in[15];
  const float* glu_d1 = (const float*)d_in[16];
  const float* glu_d2 = (const float*)d_in[17];
  const float* glu_g1 = (const float*)d_in[18];
  const float* glu_g2 = (const float*)d_in[19];
  const float* glu_g3 = (const float*)d_in[20];
  const float* dense_out_w = (const float*)d_in[21];
  float* out = (float*)d_out;

  char* ws = (char*)d_ws;
  size_t off = 0;
  auto alloc = [&](size_t bytes) -> void* {
    off = (off + 255) & ~(size_t)255;
    void* p = ws + off;
    off += bytes;
    return p;
  };
  auto packEl = [](int N, int K) -> size_t { return (size_t)(N / 16) * (K / 32) * 512; };

  __bf16* p_fd1 = (__bf16*)alloc(packEl(256, 96) * 2);
  __bf16* p_fc1 = (__bf16*)alloc(packEl(256, 768) * 2);
  __bf16* p_fc2 = (__bf16*)alloc(packEl(256, 768) * 2);
  __bf16* p_fd2 = (__bf16*)alloc(packEl(256, 256) * 2);
  __bf16* p_sd1 = (__bf16*)alloc(packEl(256, 384) * 2);
  __bf16* p_sd2 = (__bf16*)alloc(packEl(256, 256) * 2);
  __bf16* p_gd1 = (__bf16*)alloc(packEl(256, 256) * 2);
  __bf16* p_gd2 = (__bf16*)alloc(packEl(256, 256) * 2);
  __bf16* p_gg1 = (__bf16*)alloc(packEl(256, 256) * 2);
  __bf16* p_gg2 = (__bf16*)alloc(packEl(256, 256) * 2);
  __bf16* p_gg3 = (__bf16*)alloc(packEl(256, 256) * 2);
  __bf16* p_ih1 = (__bf16*)alloc(packEl(768, 256) * 2);
  __bf16* p_hh1 = (__bf16*)alloc(packEl(768, 256) * 2);
  __bf16* p_ih2 = (__bf16*)alloc(packEl(768, 256) * 2);
  __bf16* p_hh2 = (__bf16*)alloc(packEl(768, 256) * 2);
  __bf16* p_ih3 = (__bf16*)alloc(packEl(768, 256) * 2);
  __bf16* p_hh3 = (__bf16*)alloc(packEl(768, 256) * 2);
  __bf16* p_out = (__bf16*)alloc(packEl(256, 256) * 2);   // N padded 40->256
  float* Ain = (float*)alloc((size_t)256 * 68 * 96 * 4);
  float* fbuf = (float*)alloc((size_t)256 * 68 * 256 * 4);  // f, later conv2 out
  float* c1buf = (float*)alloc((size_t)256 * 66 * 256 * 4); // conv1 out, later cond
  __bf16* phase = (__bf16*)alloc((size_t)256 * 20480 * 2);

  auto pk = [&](const float* src, __bf16* dst, int N, int K, int rN, int rK, int mode) {
    long total = (long)packEl(N, K);
    pack_w<<<dim3((unsigned)((total + 255) / 256)), dim3(256), 0, stream>>>(
        src, dst, N, K, rN, rK, mode);
  };
  pk(fdense1_w, p_fd1, 256, 96, 256, 84, 0);
  pk(fconv1_w, p_fc1, 256, 768, 256, 768, 1);
  pk(fconv2_w, p_fc2, 256, 768, 256, 768, 1);
  pk(fdense2_w, p_fd2, 256, 256, 256, 256, 0);
  pk(sig_dense1_w, p_sd1, 256, 384, 256, 376, 0);
  pk(sig_dense2_w, p_sd2, 256, 256, 256, 256, 0);
  pk(glu_d1, p_gd1, 256, 256, 256, 256, 0);
  pk(glu_d2, p_gd2, 256, 256, 256, 256, 0);
  pk(glu_g1, p_gg1, 256, 256, 256, 256, 0);
  pk(glu_g2, p_gg2, 256, 256, 256, 256, 0);
  pk(glu_g3, p_gg3, 256, 256, 256, 256, 0);
  pk(gru1_wih, p_ih1, 768, 256, 768, 256, 0);
  pk(gru1_whh, p_hh1, 768, 256, 768, 256, 0);
  pk(gru2_wih, p_ih2, 768, 256, 768, 256, 0);
  pk(gru2_whh, p_hh2, 768, 256, 768, 256, 0);
  pk(gru3_wih, p_ih3, 768, 256, 768, 256, 0);
  pk(gru3_whh, p_hh3, 768, 256, 768, 256, 0);
  pk(dense_out_w, p_out, 256, 256, 40, 256, 0);

  build_ain<<<dim3((256 * 68 * 96 + 255) / 256), dim3(256), 0, stream>>>(
      features, period, pembed, Ain);
  phase_kernel<<<dim3(256), dim3(256), 0, stream>>>(period, phase);

  auto gemmBlocks = [](int M) -> unsigned {
    long tiles = (long)(M / 16) * 16;
    return (unsigned)((tiles + 7) / 8);
  };
  // f = tanh(concat(features,pembed) @ fdense1^T)            (b,68,256)
  gemm_rt<68, 96><<<dim3(gemmBlocks(256 * 68)), dim3(256), 0, stream>>>(
      Ain, 68 * 96, 96, p_fd1, fbuf, 68 * 256, 256, 256 * 68);
  // conv1: K=768 over 3 consecutive frames                   (b,66,256)
  gemm_rt<66, 768><<<dim3(gemmBlocks(256 * 66)), dim3(256), 0, stream>>>(
      fbuf, 68 * 256, 256, p_fc1, c1buf, 66 * 256, 256, 256 * 66);
  // conv2 -> reuse fbuf                                      (b,64,256)
  gemm_rt<64, 768><<<dim3(gemmBlocks(256 * 64)), dim3(256), 0, stream>>>(
      c1buf, 66 * 256, 256, p_fc2, fbuf, 64 * 256, 256, 256 * 64);
  // cond = tanh(conv2 @ fdense2^T) -> reuse c1buf            (b,64,256)
  gemm_rt<64, 256><<<dim3(gemmBlocks(256 * 64)), dim3(256), 0, stream>>>(
      fbuf, 64 * 256, 256, p_fd2, c1buf, 64 * 256, 256, 256 * 64);

  // persistent recurrent kernel: 16 workgroups x 16 batch rows
  fargan_rnn<<<dim3(16), dim3(256), 0, stream>>>(
      c1buf, phase, period, p_sd1, p_gd1, p_sd2, p_gd2, p_ih1, p_hh1, p_gg1, p_ih2,
      p_hh2, p_gg2, p_ih3, p_hh3, p_gg3, p_out, out);
}
